// TemporalFusionTransformer_27367531610307
// MI455X (gfx1250) — matrix operand order repaired and checked by reading, not compiled, not run on hardware
//
#include <hip/hip_runtime.h>

// ---------------- CDNA5 WMMA types ----------------
typedef __attribute__((ext_vector_type(16))) __bf16 v16bf;
typedef __attribute__((ext_vector_type(8)))  float  v8f;

// Problem constants
#define BB   16
#define TT   256
#define VV   64
#define FIN_ 256
#define HH   256
#define PP   8
#define NHH  8
#define DD   32
#define LL   2
#define THH  248        // T - P
#define FF_  1024       // 4*H
#define M_EMB (BB*TT*VV)
#define MHIST (BB*THH)
#define MDEC  (BB*PP)

// ---------------- fragment loaders (bf16 WMMA 16x16x32) ----------------
// A (16x32): lane L holds row L&15; lanes 16-31 take K offset +16; vgpr v:
//   K = laneHalf*16 + v*2 + (bit)
__device__ inline v16bf frag_a_lane(const __bf16* arow, int k0, int lane) {
  v16bf f;
  int koff = k0 + ((lane & 16) ? 16 : 0);
#pragma unroll
  for (int g = 0; g < 8; ++g) {
    int kk = koff + g * 2;
    f[2*g]   = arow[kk];
    f[2*g+1] = arow[kk+1];
  }
  return f;
}
__device__ inline v16bf frag_a_lane(const float* arow, int k0, int lane) {
  v16bf f;
  int koff = k0 + ((lane & 16) ? 16 : 0);
#pragma unroll
  for (int g = 0; g < 8; ++g) {
    int kk = koff + g * 2;
    f[2*g]   = (__bf16)arow[kk];
    f[2*g+1] = (__bf16)arow[kk+1];
  }
  return f;
}
// B (32x16): lane L = column L&15; lanes 16-31 take K offset +16; K contiguous.
__device__ inline v16bf frag_b_lane(const __bf16* bcol, int k0, int lane) {
  v16bf f;
  int kk = k0 + ((lane & 16) ? 16 : 0);
#pragma unroll
  for (int e = 0; e < 16; ++e) f[e] = bcol[kk + e];
  return f;
}
__device__ inline v16bf frag_b_strided(const __bf16* bcol, int strideK, int k0,
                                       int kmax, int lane) {
  v16bf f;
  int kk = k0 + ((lane & 16) ? 16 : 0);
#pragma unroll
  for (int e = 0; e < 16; ++e) {
    int k = kk + e; if (k > kmax) k = kmax;
    f[e] = bcol[(size_t)k * strideK];
  }
  return f;
}
__device__ inline v8f wmma_bf(v16bf a, v16bf b, v8f c) {
  return __builtin_amdgcn_wmma_f32_16x16x32_bf16(false, a, false, b, (short)0, c,
                                                 false, false);
}

// ---------------- gfx1250 async copy: global -> LDS, 16 bytes/lane ---------
__device__ inline void async_load16B(unsigned lds_off, const void* gaddr) {
  asm volatile("global_load_async_to_lds_b128 %0, %1, off"
               :: "v"(lds_off), "v"(gaddr) : "memory");
}

// ---------------- weight convert+transpose: dst[b][n][k] = src[b][k][n] ----
__global__ void k_convT(__bf16* dst, const float* src, int batch, int K, int N) {
  size_t i = (size_t)blockIdx.x * blockDim.x + threadIdx.x;
  size_t tot = (size_t)batch * K * N;
  if (i >= tot) return;
  int n = (int)(i % N);
  size_t j = i / N;
  int k = (int)(j % K);
  int bi = (int)(j / K);
  dst[((size_t)bi * N + n) * K + k] = (__bf16)src[((size_t)bi * K + k) * N + n];
}

__global__ void k_f2b(__bf16* dst, const float* src, size_t n) {
  size_t i = (size_t)blockIdx.x * blockDim.x + threadIdx.x;
  if (i < n) dst[i] = (__bf16)src[i];
}

// ---------------- embedding GEMM with async-to-LDS double buffering --------
// embBf[m][h] = bf16( x[m][:] @ We[:][h] + be[h] )
// Block: 16 rows x 256 cols; A tile (16x64 f32) staged in LDS via
// global_load_async_to_lds_b128, double buffered over K.
__global__ void k_embed_async(const float* __restrict__ X,
                              const __bf16* __restrict__ BT,
                              const float* __restrict__ bias,
                              __bf16* __restrict__ dst) {
  __shared__ float abuf[2][16][64];
  int wave = threadIdx.x >> 5, lane = threadIdx.x & 31;
  int m0 = blockIdx.x * 16;
  // staging: thread -> one 16B chunk of the 16x64 tile
  int srow = threadIdx.x >> 4;         // 0..15
  int sseg = (threadIdx.x & 15) * 4;   // float offset within row chunk
  const float* gsrc = X + (size_t)(m0 + srow) * FIN_ + sseg;
  // each wave computes two 16x16 tiles: cols wave*32 and wave*32+16
  int col0 = wave * 32 + (lane & 15);
  int col1 = col0 + 16;
  const __bf16* b0 = BT + (size_t)col0 * FIN_;
  const __bf16* b1 = BT + (size_t)col1 * FIN_;
  v8f acc0 = {}, acc1 = {};

  async_load16B((unsigned)(size_t)&abuf[0][srow][sseg], gsrc);
  for (int c = 0; c < 4; ++c) {      // K = 256 = 4 chunks of 64
    int k0 = c * 64;
    if (c + 1 < 4) {
      __syncthreads();  // prior reads of buf[(c+1)&1] complete
      async_load16B((unsigned)(size_t)&abuf[(c + 1) & 1][srow][sseg],
                    gsrc + k0 + 64);
      asm volatile("s_wait_asynccnt 0x1" ::: "memory");  // chunk c landed
    } else {
      asm volatile("s_wait_asynccnt 0x0" ::: "memory");
    }
    __syncthreads();    // chunk c visible to all waves
    const float* arow = &abuf[c & 1][lane & 15][0];
    v16bf fa0 = frag_a_lane(arow, 0, lane);
    acc0 = wmma_bf(fa0, frag_b_lane(b0, k0, lane), acc0);
    acc1 = wmma_bf(fa0, frag_b_lane(b1, k0, lane), acc1);
    v16bf fa1 = frag_a_lane(arow, 32, lane);
    acc0 = wmma_bf(fa1, frag_b_lane(b0, k0 + 32, lane), acc0);
    acc1 = wmma_bf(fa1, frag_b_lane(b1, k0 + 32, lane), acc1);
  }
  float bc0 = bias[col0], bc1 = bias[col1];
#pragma unroll
  for (int i = 0; i < 8; ++i) {
    int mm = m0 + i + ((lane & 16) ? 8 : 0);
    dst[(size_t)mm * HH + col0] = (__bf16)(acc0[i] + bc0);
    dst[(size_t)mm * HH + col1] = (__bf16)(acc1[i] + bc1);
  }
}

// ---------------- generic tiled GEMM: D = act(A @ B^T + bias) --------------
// Row mapping: g = (m/q)*r + (m%q)*s + o ;  A row ptr = A + g*ldg + aoff
// Each wave computes one 16x16 tile; block = 8 waves along N.
template <typename AT>
__global__ void k_gemm(const AT* __restrict__ A, int q, int r, int s, int o,
                       int ldg, int aoff, const __bf16* __restrict__ BT, int ldb,
                       const float* __restrict__ bias, void* __restrict__ dst,
                       int dstBf, int ldd, int M, int N, int K, int act) {
  int wave = threadIdx.x >> 5;
  int lane = threadIdx.x & 31;
  int m0 = blockIdx.x * 16;
  int n0 = (blockIdx.y * 8 + wave) * 16;
  if (n0 >= N) return;
  int mrow = m0 + (lane & 15);
  if (mrow >= M) mrow = M - 1;
  int g = (mrow / q) * r + (mrow % q) * s + o;
  const AT* arow = A + (size_t)g * ldg + aoff;
  int col = n0 + (lane & 15);
  int ccl = col < N ? col : N - 1;
  const __bf16* bcol = BT + (size_t)ccl * ldb;
  v8f acc = {};
  for (int k0 = 0; k0 < K; k0 += 32) {
    __builtin_prefetch(arow + k0 + 128, 0, 1);
    v16bf fa = frag_a_lane(arow, k0, lane);
    v16bf fb = frag_b_lane(bcol, k0, lane);
    acc = wmma_bf(fa, fb, acc);
  }
  float bcv = (bias && col < N) ? bias[col] : 0.f;
#pragma unroll
  for (int i = 0; i < 8; ++i) {
    int mm = m0 + i + ((lane & 16) ? 8 : 0);
    if (mm < M && col < N) {
      float v = acc[i] + bcv;
      if (act == 1) v = v > 0.f ? v : 0.f;
      else if (act == 2) v = 1.f / (1.f + __expf(-v));
      if (dstBf) ((__bf16*)dst)[(size_t)mm * ldd + col] = (__bf16)v;
      else       ((float*)dst)[(size_t)mm * ldd + col] = v;
    }
  }
}

// ---------------- per-row softmax (small N) --------------------------------
__global__ void k_softmax_rows(float* dst, const float* src, int M, int N) {
  int m = blockIdx.x * blockDim.x + threadIdx.x;
  if (m >= M) return;
  const float* rw = src + (size_t)m * N;
  float* dw = dst + (size_t)m * N;
  float mx = -1e30f;
  for (int i = 0; i < N; ++i) mx = fmaxf(mx, rw[i]);
  float sm = 0.f;
  for (int i = 0; i < N; ++i) { float e = __expf(rw[i] - mx); dw[i] = e; sm += e; }
  float inv = 1.f / sm;
  for (int i = 0; i < N; ++i) dw[i] *= inv;
}

// ---------------- GRN + layernorm + weighted accumulate (VSN core) ---------
// out[m] += w[m][n] * LN( lin*sigmoid(gate) + emb_n , g, beta )
__global__ void k_grn(const __bf16* __restrict__ emb, int q, int r, int s, int o,
                      int ldg, const __bf16* __restrict__ WT,
                      const __bf16* __restrict__ GWT, const float* __restrict__ bW,
                      const float* __restrict__ bG, const float* __restrict__ gv,
                      const float* __restrict__ bev, const float* __restrict__ wsm,
                      float* __restrict__ out, int M, int nvar) {
  const int Hn = HH;
  int n = blockIdx.y;
  int m0 = blockIdx.x * 16;
  int wave = threadIdx.x >> 5, lane = threadIdx.x & 31;
  __shared__ float sval[16][HH];
  __shared__ float smu[16], srs[16];
  const __bf16* Wn = WT + (size_t)n * Hn * Hn;
  const __bf16* Gn = GWT + (size_t)n * Hn * Hn;
  int mrow = m0 + (lane & 15);
  if (mrow >= M) mrow = M - 1;
  int gg = (mrow / q) * r + (mrow % q) * s + o;
  const __bf16* arow = emb + (size_t)gg * ldg + n * Hn;
  int col0 = wave * 32 + (lane & 15);
  int col1 = col0 + 16;
  const __bf16* b0w = Wn + (size_t)col0 * Hn;
  const __bf16* b1w = Wn + (size_t)col1 * Hn;
  const __bf16* b0g = Gn + (size_t)col0 * Hn;
  const __bf16* b1g = Gn + (size_t)col1 * Hn;
  v8f l0 = {}, l1 = {}, a0 = {}, a1 = {};
  for (int k0 = 0; k0 < Hn; k0 += 32) {
    v16bf fa = frag_a_lane(arow, k0, lane);
    l0 = wmma_bf(fa, frag_b_lane(b0w, k0, lane), l0);
    l1 = wmma_bf(fa, frag_b_lane(b1w, k0, lane), l1);
    a0 = wmma_bf(fa, frag_b_lane(b0g, k0, lane), a0);
    a1 = wmma_bf(fa, frag_b_lane(b1g, k0, lane), a1);
  }
  const float* bWn = bW + (size_t)n * Hn;
  const float* bGn = bG + (size_t)n * Hn;
#pragma unroll
  for (int i = 0; i < 8; ++i) {
    int rr = i + ((lane & 16) ? 8 : 0);
    int mm = m0 + rr; int mmc = mm < M ? mm : M - 1;
    int ge = (mmc / q) * r + (mmc % q) * s + o;
    const __bf16* er = emb + (size_t)ge * ldg + n * Hn;
    float s0 = 1.f / (1.f + __expf(-(a0[i] + bGn[col0])));
    sval[rr][col0] = (l0[i] + bWn[col0]) * s0 + (float)er[col0];
    float s1 = 1.f / (1.f + __expf(-(a1[i] + bGn[col1])));
    sval[rr][col1] = (l1[i] + bWn[col1]) * s1 + (float)er[col1];
  }
  __syncthreads();
  if (threadIdx.x < 16) {
    int rr = threadIdx.x;
    float sm = 0.f;
    for (int c = 0; c < Hn; ++c) sm += sval[rr][c];
    float mu = sm / Hn;
    float v2 = 0.f;
    for (int c = 0; c < Hn; ++c) { float d = sval[rr][c] - mu; v2 += d * d; }
    smu[rr] = mu;
    srs[rr] = rsqrtf(v2 / Hn + 1e-5f);
  }
  __syncthreads();
  int rr = threadIdx.x >> 4;
  int cb = (threadIdx.x & 15) * 16;
  int mm = m0 + rr;
  if (mm < M) {
    float wn = wsm[(size_t)mm * nvar + n];
    const float* gvn = gv + (size_t)n * Hn;
    const float* ben = bev + (size_t)n * Hn;
    for (int c = cb; c < cb + 16; ++c) {
      float ov = (sval[rr][c] - smu[rr]) * srs[rr] * gvn[c] + ben[c];
      atomicAdd(&out[(size_t)mm * Hn + c], ov * wn);
    }
  }
}

// ---------------- attention scores + softmax (K = head_dim = 32) -----------
__global__ void k_scores(const __bf16* __restrict__ Qb, const __bf16* __restrict__ Kb,
                         __bf16* __restrict__ Sout, float* __restrict__ aw,
                         int Tq, int Kpad, float scale) {
  int b = blockIdx.z, h = blockIdx.y, q0 = blockIdx.x * 16;
  int wave = threadIdx.x >> 5, lane = threadIdx.x & 31;
  __shared__ float sc[16][256];
  int qr = q0 + (lane & 15); if (qr >= Tq) qr = Tq - 1;
  const __bf16* arow = Qb + ((size_t)(b * Tq + qr)) * HH + h * DD;
  int ntiles = Kpad / 16;
  for (int t = wave; t < ntiles; t += 8) {
    int kc = t * 16 + (lane & 15);
    int kcc = kc < Tq ? kc : Tq - 1;
    const __bf16* bcol = Kb + ((size_t)(b * Tq + kcc)) * HH + h * DD;
    v8f acc = {};
    v16bf fa = frag_a_lane(arow, 0, lane);
    v16bf fb = frag_b_lane(bcol, 0, lane);
    acc = wmma_bf(fa, fb, acc);
#pragma unroll
    for (int i = 0; i < 8; ++i) {
      int rr = i + ((lane & 16) ? 8 : 0);
      sc[rr][t * 16 + (lane & 15)] = acc[i] * scale;
    }
  }
  __syncthreads();
  if (threadIdx.x < 16) {
    int rw = threadIdx.x;
    int qq = q0 + rw;
    if (qq < Tq) {
      float mx = -1e30f;
      for (int k = 0; k < Tq; ++k) mx = fmaxf(mx, sc[rw][k]);
      float sm = 0.f;
      for (int k = 0; k < Tq; ++k) { float e = __expf(sc[rw][k] - mx); sc[rw][k] = e; sm += e; }
      float inv = 1.f / sm;
      size_t base = ((size_t)((b * NHH + h) * Tq) + qq) * Kpad;
      for (int k = 0; k < Kpad; ++k)
        Sout[base + k] = (__bf16)((k < Tq) ? sc[rw][k] * inv : 0.f);
      if (aw) {
        size_t ab = ((size_t)((b * NHH + h) * Tq) + qq) * Tq;
        for (int k = 0; k < Tq; ++k) aw[ab + k] = sc[rw][k] * inv;
      }
    }
  }
}

// ---------------- ctx = softmax(scores) @ V (block = 2 waves, N = 32) ------
__global__ void k_ctx(const __bf16* __restrict__ Sb, const __bf16* __restrict__ Vb,
                      float* __restrict__ ctx, int Tq, int Kpad) {
  int b = blockIdx.z, h = blockIdx.y, q0 = blockIdx.x * 16;
  int wave = threadIdx.x >> 5, lane = threadIdx.x & 31;
  int qr = q0 + (lane & 15); if (qr >= Tq) qr = Tq - 1;
  const __bf16* arow = Sb + ((size_t)((b * NHH + h) * Tq) + qr) * Kpad;
  int dcol = wave * 16 + (lane & 15);
  const __bf16* bcol = Vb + ((size_t)(b * Tq)) * HH + h * DD + dcol;
  v8f acc = {};
  for (int k0 = 0; k0 < Kpad; k0 += 32) {
    v16bf fa = frag_a_lane(arow, k0, lane);
    v16bf fb = frag_b_strided(bcol, HH, k0, Tq - 1, lane);
    acc = wmma_bf(fa, fb, acc);
  }
#pragma unroll
  for (int i = 0; i < 8; ++i) {
    int qq = q0 + i + ((lane & 16) ? 8 : 0);
    if (qq < Tq) ctx[((size_t)(b * Tq + qq)) * HH + h * DD + dcol] = acc[i];
  }
}

// ---------------- residual + layernorm -------------------------------------
__global__ void k_addln(float* dst, const float* a, const float* b,
                        const float* g, const float* be) {
  int row = blockIdx.x, tid = threadIdx.x;
  __shared__ float red[HH];
  size_t idx = (size_t)row * HH + tid;
  float v = a[idx] + b[idx];
  red[tid] = v; __syncthreads();
  for (int s2 = 128; s2 > 0; s2 >>= 1) { if (tid < s2) red[tid] += red[tid + s2]; __syncthreads(); }
  float mu = red[0] / HH; __syncthreads();
  float d = v - mu; red[tid] = d * d; __syncthreads();
  for (int s2 = 128; s2 > 0; s2 >>= 1) { if (tid < s2) red[tid] += red[tid + s2]; __syncthreads(); }
  float rstd = rsqrtf(red[0] / HH + 1e-5f);
  dst[idx] = (v - mu) * rstd * g[tid] + be[tid];
}

// final: LN(dec*gate + fut)
__global__ void k_ln2(float* dst, const float* dec, const float* fut,
                      const float* gate, const float* g, const float* be) {
  int row = blockIdx.x, tid = threadIdx.x;
  int b = row / PP;
  __shared__ float red[HH];
  size_t idx = (size_t)row * HH + tid;
  float v = dec[idx] * gate[(size_t)b * HH + tid] + fut[idx];
  red[tid] = v; __syncthreads();
  for (int s2 = 128; s2 > 0; s2 >>= 1) { if (tid < s2) red[tid] += red[tid + s2]; __syncthreads(); }
  float mu = red[0] / HH; __syncthreads();
  float d = v - mu; red[tid] = d * d; __syncthreads();
  for (int s2 = 128; s2 > 0; s2 >>= 1) { if (tid < s2) red[tid] += red[tid + s2]; __syncthreads(); }
  float rstd = rsqrtf(red[0] / HH + 1e-5f);
  dst[idx] = (v - mu) * rstd * g[tid] + be[tid];
}

// concat(static_ctx[:, -1], enc[:, -1]) -> bf16 [B][2H]
__global__ void k_concat(__bf16* dst, const float* sctx, const float* X) {
  int i = blockIdx.x * blockDim.x + threadIdx.x;
  if (i >= BB * 2 * HH) return;
  int b = i >> 9, c = i & 511;
  float v = (c < HH) ? sctx[(size_t)b * HH + c]
                     : X[((size_t)(b * THH + THH - 1)) * HH + (c - HH)];
  dst[i] = (__bf16)v;
}

// dec input: fut_enc + gate * enc_ctx
__global__ void k_decin(float* dX, __bf16* dXb, const float* fut,
                        const float* gate, const float* X) {
  int i = blockIdx.x * blockDim.x + threadIdx.x;
  if (i >= MDEC * HH) return;
  int h = i & 255;
  int b = (i >> 8) / PP;
  float v = fut[i] + gate[(size_t)b * HH + h] * X[((size_t)(b * THH + THH - 1)) * HH + h];
  dX[i] = v; dXb[i] = (__bf16)v;
}

// preds = out @ Wo + bo  (N=1 -> block reduction)
__global__ void k_pred(float* out, const float* X, const float* Wo, const float* bo) {
  int row = blockIdx.x, tid = threadIdx.x;
  __shared__ float red[HH];
  red[tid] = X[(size_t)row * HH + tid] * Wo[tid];
  __syncthreads();
  for (int s2 = 128; s2 > 0; s2 >>= 1) { if (tid < s2) red[tid] += red[tid + s2]; __syncthreads(); }
  if (tid == 0) out[row] = red[0] + bo[0];
}

// ---------------- host side -------------------------------------------------
enum {
  IN_X = 0, IN_WE = 1, IN_BE = 2,
  ST_SWW = 3, ST_SWB = 4, ST_W = 5, ST_B = 6, ST_GW = 7, ST_GB = 8, ST_G = 9, ST_BETA = 10,
  HI_SWW = 11, HI_SWB = 12, HI_W = 13, HI_B = 14, HI_GW = 15, HI_GB = 16, HI_G = 17, HI_BETA = 18,
  FU_SWW = 19, FU_SWB = 20, FU_W = 21, FU_B = 22, FU_GW = 23, FU_GB = 24, FU_G = 25, FU_BETA = 26,
  ENC0 = 27, ENC1 = 35, DEC0 = 43, DEC1 = 51,
  PW0 = 59,
  IN_WG = 75, IN_BG = 76, IN_LNG = 77, IN_LNB = 78, IN_WO = 79, IN_BO = 80
};

static inline size_t al256(size_t x) { return (x + 255) & ~(size_t)255; }

extern "C" void kernel_launch(void* const* d_in, const int* in_sizes, int n_in,
                              void* d_out, int out_size, void* d_ws, size_t ws_size,
                              hipStream_t stream) {
  (void)in_sizes; (void)n_in; (void)out_size; (void)ws_size;
  const float* fin[96];
  for (int i = 0; i < 81; ++i) fin[i] = (const float*)d_in[i];
  float* out = (float*)d_out;
  float* aw_out = out + 128;   // preds first, then attention weights

  char* ws = (char*)d_ws;
  size_t off = 0;
  auto A = [&](size_t bytes) { void* p = ws + off; off = al256(off + bytes); return p; };

  // bf16 weight copies (transposed [N][K])
  __bf16* embBf  = (__bf16*)A((size_t)M_EMB * HH * 2);
  __bf16* WeT    = (__bf16*)A((size_t)HH * FIN_ * 2);
  __bf16* swHT   = (__bf16*)A((size_t)64 * 16384 * 2);
  __bf16* grnHW  = (__bf16*)A((size_t)64 * HH * HH * 2);
  __bf16* grnHG  = (__bf16*)A((size_t)64 * HH * HH * 2);
  __bf16* swST   = (__bf16*)A((size_t)4 * 1024 * 2);
  __bf16* grnSW  = (__bf16*)A((size_t)4 * HH * HH * 2);
  __bf16* grnSG  = (__bf16*)A((size_t)4 * HH * HH * 2);
  __bf16* swFT   = (__bf16*)A((size_t)8 * 2048 * 2);
  __bf16* grnFW  = (__bf16*)A((size_t)8 * HH * HH * 2);
  __bf16* grnFG  = (__bf16*)A((size_t)8 * HH * HH * 2);
  __bf16* attWT[4][3];           // [enc0,enc1,dec0,dec1][q,k,v]
  for (int l = 0; l < 4; ++l)
    for (int m = 0; m < 3; ++m) attWT[l][m] = (__bf16*)A((size_t)HH * HH * 2);
  __bf16* W1T[4]; __bf16* W2T[4];
  for (int i = 0; i < 4; ++i) {
    W1T[i] = (__bf16*)A((size_t)FF_ * HH * 2);
    W2T[i] = (__bf16*)A((size_t)HH * FF_ * 2);
  }
  __bf16* WgT = (__bf16*)A((size_t)HH * 2 * HH * 2);

  // activations
  float*  logitsH = (float*)A((size_t)MHIST * 64 * 4);
  float*  wH      = (float*)A((size_t)MHIST * 64 * 4);
  float*  XA      = (float*)A((size_t)MHIST * HH * 4);   // hist_enc / enc ping
  float*  XB      = (float*)A((size_t)MHIST * HH * 4);
  float*  ctxF    = (float*)A((size_t)MHIST * HH * 4);
  __bf16* Xbf     = (__bf16*)A((size_t)MHIST * HH * 2);
  __bf16* Qbf     = (__bf16*)A((size_t)MHIST * HH * 2);
  __bf16* Kbf     = (__bf16*)A((size_t)MHIST * HH * 2);
  __bf16* Vbf     = (__bf16*)A((size_t)MHIST * HH * 2);
  __bf16* Sbf     = (__bf16*)A((size_t)BB * NHH * THH * 256 * 2);
  __bf16* hidBf   = (__bf16*)A((size_t)MHIST * FF_ * 2);
  float*  logitsS = (float*)A((size_t)BB * 4 * 4);
  float*  wS      = (float*)A((size_t)BB * 4 * 4);
  float*  sctx    = (float*)A((size_t)BB * HH * 4);
  float*  logitsF = (float*)A((size_t)MDEC * 8 * 4);
  float*  wF      = (float*)A((size_t)MDEC * 8 * 4);
  float*  futE    = (float*)A((size_t)MDEC * HH * 4);
  __bf16* catBf   = (__bf16*)A((size_t)BB * 2 * HH * 2);
  float*  gate    = (float*)A((size_t)BB * HH * 4);
  float*  dXA     = (float*)A((size_t)MDEC * HH * 4);
  float*  dXB     = (float*)A((size_t)MDEC * HH * 4);
  float*  dctx    = (float*)A((size_t)MDEC * HH * 4);
  __bf16* dbf     = (__bf16*)A((size_t)MDEC * HH * 2);
  __bf16* dQ      = (__bf16*)A((size_t)MDEC * HH * 2);
  __bf16* dK      = (__bf16*)A((size_t)MDEC * HH * 2);
  __bf16* dV      = (__bf16*)A((size_t)MDEC * HH * 2);
  __bf16* dS      = (__bf16*)A((size_t)BB * NHH * PP * 32 * 2);
  __bf16* dhid    = (__bf16*)A((size_t)MDEC * FF_ * 2);
  float*  outLN   = (float*)A((size_t)MDEC * HH * 4);

  const int CONT = 0x40000000;  // contiguous row mapping: q huge, s=1
  dim3 blk(256);
  auto cgrid = [](size_t n) { return dim3((unsigned)((n + 255) / 256)); };

  // ---- weight prep ----
  k_convT<<<cgrid((size_t)FIN_*HH), blk, 0, stream>>>(WeT,  fin[IN_WE], 1, FIN_, HH);
  k_convT<<<cgrid((size_t)16384*64), blk, 0, stream>>>(swHT, fin[HI_SWW], 1, 16384, 64);
  k_convT<<<cgrid((size_t)64*HH*HH), blk, 0, stream>>>(grnHW, fin[HI_W], 64, HH, HH);
  k_convT<<<cgrid((size_t)64*HH*HH), blk, 0, stream>>>(grnHG, fin[HI_GW], 64, HH, HH);
  k_convT<<<cgrid((size_t)1024*4),  blk, 0, stream>>>(swST, fin[ST_SWW], 1, 1024, 4);
  k_convT<<<cgrid((size_t)4*HH*HH), blk, 0, stream>>>(grnSW, fin[ST_W], 4, HH, HH);
  k_convT<<<cgrid((size_t)4*HH*HH), blk, 0, stream>>>(grnSG, fin[ST_GW], 4, HH, HH);
  k_convT<<<cgrid((size_t)2048*8),  blk, 0, stream>>>(swFT, fin[FU_SWW], 1, 2048, 8);
  k_convT<<<cgrid((size_t)8*HH*HH), blk, 0, stream>>>(grnFW, fin[FU_W], 8, HH, HH);
  k_convT<<<cgrid((size_t)8*HH*HH), blk, 0, stream>>>(grnFG, fin[FU_GW], 8, HH, HH);
  int attBase[4] = {ENC0, ENC1, DEC0, DEC1};
  for (int l = 0; l < 4; ++l)
    for (int m = 0; m < 3; ++m)
      k_convT<<<cgrid((size_t)HH*HH), blk, 0, stream>>>(attWT[l][m],
          fin[attBase[l] + 2*m], 1, HH, HH);
  for (int i = 0; i < 4; ++i) {
    k_convT<<<cgrid((size_t)HH*FF_), blk, 0, stream>>>(W1T[i], fin[PW0 + 4*i + 0], 1, HH, FF_);
    k_convT<<<cgrid((size_t)FF_*HH), blk, 0, stream>>>(W2T[i], fin[PW0 + 4*i + 2], 1, FF_, HH);
  }
  k_convT<<<cgrid((size_t)2*HH*HH), blk, 0, stream>>>(WgT, fin[IN_WG], 1, 2*HH, HH);

  // zero atomic accumulators
  hipMemsetAsync(XA,   0, (size_t)MHIST * HH * 4, stream);
  hipMemsetAsync(sctx, 0, (size_t)BB * HH * 4, stream);
  hipMemsetAsync(futE, 0, (size_t)MDEC * HH * 4, stream);

  // ---- embedding: embBf = bf16(x @ We + be), async-LDS staged A ----
  k_embed_async<<<dim3(M_EMB/16), blk, 0, stream>>>(
      fin[IN_X], WeT, fin[IN_BE], embBf);

  // ---- hist VSN ----
  k_gemm<__bf16><<<dim3(MHIST/16, 1), blk, 0, stream>>>(
      embBf, THH, TT, 1, 0, VV*HH, 0, swHT, 16384, fin[HI_SWB],
      logitsH, 0, 64, MHIST, 64, 16384, 0);
  k_softmax_rows<<<cgrid(MHIST), blk, 0, stream>>>(wH, logitsH, MHIST, 64);
  k_grn<<<dim3(MHIST/16, 64), blk, 0, stream>>>(
      embBf, THH, TT, 1, 0, VV*HH, grnHW, grnHG, fin[HI_B], fin[HI_GB],
      fin[HI_G], fin[HI_BETA], wH, XA, MHIST, 64);

  // ---- static VSN (only t = T-1 needed) ----
  k_gemm<__bf16><<<dim3(1, 1), blk, 0, stream>>>(
      embBf, 1, TT, 1, TT-1, VV*HH, 0, swST, 1024, fin[ST_SWB],
      logitsS, 0, 4, BB, 4, 1024, 0);
  k_softmax_rows<<<cgrid(BB), blk, 0, stream>>>(wS, logitsS, BB, 4);
  k_grn<<<dim3(1, 4), blk, 0, stream>>>(
      embBf, 1, TT, 1, TT-1, VV*HH, grnSW, grnSG, fin[ST_B], fin[ST_GB],
      fin[ST_G], fin[ST_BETA], wS, sctx, BB, 4);

  // ---- future VSN (t in [T-P, T), first P variables) ----
  k_gemm<__bf16><<<dim3(MDEC/16, 1), blk, 0, stream>>>(
      embBf, PP, TT, 1, THH, VV*HH, 0, swFT, 2048, fin[FU_SWB],
      logitsF, 0, 8, MDEC, 8, 2048, 0);
  k_softmax_rows<<<cgrid(MDEC), blk, 0, stream>>>(wF, logitsF, MDEC, 8);
  k_grn<<<dim3(MDEC/16, 8), blk, 0, stream>>>(
      embBf, PP, TT, 1, THH, VV*HH, grnFW, grnFG, fin[FU_B], fin[FU_GB],
      fin[FU_G], fin[FU_BETA], wF, futE, MDEC, 8);

  // ---- encoder ----
  const float scale = 0.17677669529663687f;  // 1/sqrt(32)
  for (int l = 0; l < LL; ++l) {
    int ab = attBase[l];
    k_f2b<<<cgrid((size_t)MHIST*HH), blk, 0, stream>>>(Xbf, XA, (size_t)MHIST*HH);
    __bf16* qkv[3] = {Qbf, Kbf, Vbf};
    for (int m = 0; m < 3; ++m)
      k_gemm<__bf16><<<dim3(MHIST/16, 2), blk, 0, stream>>>(
          Xbf, CONT, 0, 1, 0, HH, 0, attWT[l][m], HH, fin[ab + 2*m + 1],
          qkv[m], 1, HH, MHIST, HH, HH, 0);
    k_scores<<<dim3(16, NHH, BB), blk, 0, stream>>>(
        Qbf, Kbf, Sbf, (l == LL-1) ? aw_out : nullptr, THH, 256, scale);
    k_ctx<<<dim3(16, NHH, BB), dim3(64), 0, stream>>>(Sbf, Vbf, ctxF, THH, 256);
    k_addln<<<dim3(MHIST), blk, 0, stream>>>(XB, ctxF, XA, fin[ab+6], fin[ab+7]);
    k_f2b<<<cgrid((size_t)MHIST*HH), blk, 0, stream>>>(Xbf, XB, (size_t)MHIST*HH);
    k_gemm<__bf16><<<dim3(MHIST/16, 8), blk, 0, stream>>>(
        Xbf, CONT, 0, 1, 0, HH, 0, W1T[l], HH, fin[PW0 + 4*l + 1],
        hidBf, 1, FF_, MHIST, FF_, HH, 1);
    k_gemm<__bf16><<<dim3(MHIST/16, 2), blk, 0, stream>>>(
        hidBf, CONT, 0, 1, 0, FF_, 0, W2T[l], FF_, fin[PW0 + 4*l + 3],
        XA, 0, HH, MHIST, HH, FF_, 0);
  }

  // ---- gate + decoder input ----
  k_concat<<<cgrid((size_t)BB*2*HH), blk, 0, stream>>>(catBf, sctx, XA);
  k_gemm<__bf16><<<dim3(1, 2), blk, 0, stream>>>(
      catBf, CONT, 0, 1, 0, 2*HH, 0, WgT, 2*HH, fin[IN_BG],
      gate, 0, HH, BB, HH, 2*HH, 2);
  k_decin<<<cgrid((size_t)MDEC*HH), blk, 0, stream>>>(dXA, dbf, futE, gate, XA);

  // ---- decoder ----
  for (int l = 0; l < LL; ++l) {
    int ab = attBase[2 + l];
    k_f2b<<<cgrid((size_t)MDEC*HH), blk, 0, stream>>>(dbf, dXA, (size_t)MDEC*HH);
    __bf16* qkv[3] = {dQ, dK, dV};
    for (int m = 0; m < 3; ++m)
      k_gemm<__bf16><<<dim3(MDEC/16, 2), blk, 0, stream>>>(
          dbf, CONT, 0, 1, 0, HH, 0, attWT[2+l][m], HH, fin[ab + 2*m + 1],
          qkv[m], 1, HH, MDEC, HH, HH, 0);
    k_scores<<<dim3(1, NHH, BB), blk, 0, stream>>>(dQ, dK, dS, nullptr, PP, 32, scale);
    k_ctx<<<dim3(1, NHH, BB), dim3(64), 0, stream>>>(dS, dV, dctx, PP, 32);
    k_addln<<<dim3(MDEC), blk, 0, stream>>>(dXB, dctx, dXA, fin[ab+6], fin[ab+7]);
    k_f2b<<<cgrid((size_t)MDEC*HH), blk, 0, stream>>>(dbf, dXB, (size_t)MDEC*HH);
    k_gemm<__bf16><<<dim3(MDEC/16, 8), blk, 0, stream>>>(
        dbf, CONT, 0, 1, 0, HH, 0, W1T[2+l], HH, fin[PW0 + 4*(2+l) + 1],
        dhid, 1, FF_, MDEC, FF_, HH, 1);
    k_gemm<__bf16><<<dim3(MDEC/16, 2), blk, 0, stream>>>(
        dhid, CONT, 0, 1, 0, FF_, 0, W2T[2+l], FF_, fin[PW0 + 4*(2+l) + 3],
        dXA, 0, HH, MDEC, HH, FF_, 0);
  }

  // ---- final LN + output projection ----
  k_ln2<<<dim3(MDEC), blk, 0, stream>>>(outLN, dXA, futE, gate, fin[IN_LNG], fin[IN_LNB]);
  k_pred<<<dim3(MDEC), blk, 0, stream>>>(out, outLN, fin[IN_WO], fin[IN_BO]);
}
